// GATv2_88356067213530
// MI455X (gfx1250) — matrix operand order repaired and checked by reading, not compiled, not run on hardware
//
#include <hip/hip_runtime.h>

typedef __attribute__((ext_vector_type(16))) _Float16 v16h;
typedef __attribute__((ext_vector_type(8)))  _Float16 v8h;
typedef __attribute__((ext_vector_type(8)))  float    v8f;
typedef int v4i_vs __attribute__((vector_size(16)));   // matches builtin param type

#define IN_F      128
#define F1        128   // HEADS*HID
#define OUTC      64
#define EDIM      32
#define NEG_SLOPE 0.2f

// ---- optional CDNA5 async global->LDS staging (ASYNCcnt path), safely gated ----
#if defined(__HIP_DEVICE_COMPILE__) && \
    __has_builtin(__builtin_amdgcn_global_load_async_to_lds_b128) && \
    __has_builtin(__builtin_amdgcn_s_wait_asynccnt)
#define USE_ASYNC_LDS 1
#else
#define USE_ASYNC_LDS 0
#endif

// stage nfloats (multiple of 4) from global to LDS; call from all 256 threads,
// follow with __syncthreads() at the call site.
__device__ __forceinline__ void stage_lds_f32(float* __restrict__ sdst,
                                              const float* __restrict__ gsrc,
                                              int nfloats, int tid) {
#if USE_ASYNC_LDS
  for (int i = tid * 4; i < nfloats; i += 256 * 4) {
    __builtin_amdgcn_global_load_async_to_lds_b128(
        (__attribute__((address_space(1))) v4i_vs*)(gsrc + i),
        (__attribute__((address_space(3))) v4i_vs*)(sdst + i), 0, 0);
  }
  __builtin_amdgcn_s_wait_asynccnt(0);
#else
  for (int i = tid; i < nfloats; i += 256) sdst[i] = gsrc[i];
#endif
}

__device__ __forceinline__ void atomicMaxFloat(float* addr, float val) {
  // valid with init = -inf: positives via signed max, negatives via unsigned min
  if (val >= 0.0f) atomicMax((int*)addr, __float_as_int(val));
  else             atomicMin((unsigned int*)addr, __float_as_uint(val));
}

// ---------------- init kernels ----------------

__global__ void k_fill(float* __restrict__ p, float v, long long n) {
  long long i = (long long)blockIdx.x * blockDim.x + threadIdx.x;
  if (i < n) p[i] = v;
}

__global__ void k_init_out(float4* __restrict__ out, const float4* __restrict__ bias4,
                           long long n4, int F4) {
  long long i = (long long)blockIdx.x * blockDim.x + threadIdx.x;
  if (i < n4) out[i] = bias4[(int)(i % F4)];
}

// ---------------- self-loop attr = segment-mean(edge_attr by dst) ----------------

__global__ void k_loop_sums(const int* __restrict__ dstE, const float4* __restrict__ ea4,
                            float* __restrict__ deg, float* __restrict__ loopA, int E) {
  long long i = (long long)blockIdx.x * blockDim.x + threadIdx.x;  // thread per (edge, quad)
  if (i >= (long long)E * (EDIM / 4)) return;
  const int e = (int)(i >> 3), q = (int)(i & 7);
  const int d = dstE[e];
  const float4 v = ea4[i];
  float* t = &loopA[(long long)d * EDIM + q * 4];
  atomicAdd(t + 0, v.x); atomicAdd(t + 1, v.y);
  atomicAdd(t + 2, v.z); atomicAdd(t + 3, v.w);
  if (q == 0) atomicAdd(&deg[d], 1.0f);
}

__global__ void k_loop_fin(float4* __restrict__ loopA4, const float* __restrict__ deg, int N) {
  long long i = (long long)blockIdx.x * blockDim.x + threadIdx.x;
  if (i >= (long long)N * (EDIM / 4)) return;
  const float r = 1.0f / fmaxf(deg[(int)(i >> 3)], 1.0f);
  float4 v = loopA4[i];
  v.x *= r; v.y *= r; v.z *= r; v.w *= r;
  loopA4[i] = v;
}

// ---------------- dense node projection via WMMA f16 ----------------
// C[M,NCOLS] = A[M,128] @ W[128,NCOLS] + bias. Block = 8 waves; W staged once per
// block into LDS transposed as f16; each wave owns one 16-row M tile, keeps its A
// fragments in registers across all NCOLS/16 column tiles -> 4*(NCOLS/16) WMMAs.

template <int NCOLS>
__global__ __launch_bounds__(256)
void k_wmma_gemm_bias(const float* __restrict__ A, const float* __restrict__ W,
                      const float* __restrict__ bias, float* __restrict__ C, int M) {
  constexpr int K    = 128;
  constexpr int LSTR = K + 8;                     // padded row stride (halfs)
  __shared__ _Float16 sWt[NCOLS * LSTR];          // transposed: [col][k]
  for (int i = threadIdx.x; i < K * NCOLS; i += 256) {
    const int k = i / NCOLS, c = i % NCOLS;       // coalesced global read of W[k][c]
    sWt[c * LSTR + k] = (_Float16)W[i];
  }
  __syncthreads();

  const int wave = threadIdx.x >> 5, lane = threadIdx.x & 31;
  const int mt = blockIdx.x * 8 + wave;
  if (mt * 16 >= M) return;                       // no barriers after this point
  const int half16 = lane >> 4;                   // 0: lanes 0-15, 1: lanes 16-31
  const int kA = half16 * 8;                      // ISA 16-bit A-fragment K offset
  const int kB = half16 * 8;                      // ISA 16-bit B-fragment K offset
  int row = mt * 16 + (lane & 15);
  if (row >= M) row = M - 1;                      // clamp; stores guarded below

  // A fragments for the 4 k-steps, loaded once as b128s
  v16h afr[4];
  const float* Ar = A + (size_t)row * K;
#pragma unroll
  for (int ks = 0; ks < 4; ++ks) {
    const int kb = ks * 32;
    const float4 p0 = *(const float4*)(Ar + kb + kA);
    const float4 p1 = *(const float4*)(Ar + kb + kA + 4);
    const float4 q0 = *(const float4*)(Ar + kb + 16 + kA);
    const float4 q1 = *(const float4*)(Ar + kb + 16 + kA + 4);
    afr[ks][0]  = (_Float16)p0.x; afr[ks][1]  = (_Float16)p0.y;
    afr[ks][2]  = (_Float16)p0.z; afr[ks][3]  = (_Float16)p0.w;
    afr[ks][4]  = (_Float16)p1.x; afr[ks][5]  = (_Float16)p1.y;
    afr[ks][6]  = (_Float16)p1.z; afr[ks][7]  = (_Float16)p1.w;
    afr[ks][8]  = (_Float16)q0.x; afr[ks][9]  = (_Float16)q0.y;
    afr[ks][10] = (_Float16)q0.z; afr[ks][11] = (_Float16)q0.w;
    afr[ks][12] = (_Float16)q1.x; afr[ks][13] = (_Float16)q1.y;
    afr[ks][14] = (_Float16)q1.z; afr[ks][15] = (_Float16)q1.w;
  }

  const int mbase = mt * 16 + half16 * 8;         // C/D layout: 8 rows per lane half
#pragma unroll
  for (int tn = 0; tn < NCOLS / 16; ++tn) {
    const int col = tn * 16 + (lane & 15);
    v8f acc = {};
#pragma unroll
    for (int ks = 0; ks < 4; ++ks) {
      const _Float16* bp = &sWt[col * LSTR + ks * 32 + kB];
      const v8h blo = *(const v8h*)bp;            // ds_load_b128
      const v8h bhi = *(const v8h*)(bp + 16);     // ds_load_b128
      v16h b;
#pragma unroll
      for (int j = 0; j < 8; ++j) { b[j] = blo[j]; b[j + 8] = bhi[j]; }
      acc = __builtin_amdgcn_wmma_f32_16x16x32_f16(false, afr[ks], false, b,
                                                   (short)0, acc, false, false);
    }
    const float bv = bias[col];
#pragma unroll
    for (int r = 0; r < 8; ++r) {
      const int m = mbase + r;
      if (m < M) C[(size_t)m * NCOLS + col] = acc[r] + bv;
    }
  }
}

// ---------------- layer-1 edge logits ----------------
// wave per edge; lane owns 4 of 128 channels (float4 path); We1 staged in LDS.

__global__ __launch_bounds__(256)
void k_edge_logits1(const int* __restrict__ srcE, const int* __restrict__ dstE,
                    const float* __restrict__ edge_attr, const float* __restrict__ loopA,
                    const float* __restrict__ We, const float* __restrict__ att,
                    const float* __restrict__ xl, const float* __restrict__ xr,
                    float* __restrict__ elog, float* __restrict__ lmax, int E, int N) {
  __shared__ float sWe[EDIM * F1];
  stage_lds_f32(sWe, We, EDIM * F1, threadIdx.x);
  __syncthreads();

  const int wave = threadIdx.x >> 5, lane = threadIdx.x & 31;
  const long long Etot = (long long)E + N;
  const long long e = (long long)blockIdx.x * 8 + wave;
  if (e >= Etot) return;

  int s, d; const float* eap;
  if (e < E) { s = srcE[e]; d = dstE[e]; eap = edge_attr + e * EDIM; }
  else       { s = d = (int)(e - E);     eap = loopA + (long long)s * EDIM; }

  float eav[EDIM];
#pragma unroll
  for (int k4 = 0; k4 < EDIM / 4; ++k4) {
    const float4 t = ((const float4*)eap)[k4];
    eav[4 * k4 + 0] = t.x; eav[4 * k4 + 1] = t.y;
    eav[4 * k4 + 2] = t.z; eav[4 * k4 + 3] = t.w;
  }

  const int c0 = lane * 4;
  const float4 am  = *(const float4*)(att + c0);
  const float4 xlv = *(const float4*)(xl + (long long)s * F1 + c0);
  const float4 xrv = *(const float4*)(xr + (long long)d * F1 + c0);
  float4 ep = {0.f, 0.f, 0.f, 0.f};
#pragma unroll
  for (int k = 0; k < EDIM; ++k) {
    const float4 w = *(const float4*)(sWe + k * F1 + c0);   // ds_load_b128
    ep.x += eav[k] * w.x; ep.y += eav[k] * w.y;
    ep.z += eav[k] * w.z; ep.w += eav[k] * w.w;
  }
  float p = 0.0f, m;
  m = xlv.x + xrv.x + ep.x; m = (m > 0.f) ? m : NEG_SLOPE * m; p += m * am.x;
  m = xlv.y + xrv.y + ep.y; m = (m > 0.f) ? m : NEG_SLOPE * m; p += m * am.y;
  m = xlv.z + xrv.z + ep.z; m = (m > 0.f) ? m : NEG_SLOPE * m; p += m * am.z;
  m = xlv.w + xrv.w + ep.w; m = (m > 0.f) ? m : NEG_SLOPE * m; p += m * am.w;

  // reduce within each 16-lane half: lanes 0-15 = head 0, lanes 16-31 = head 1
  p += __shfl_xor(p, 8, 16);
  p += __shfl_xor(p, 4, 16);
  p += __shfl_xor(p, 2, 16);
  p += __shfl_xor(p, 1, 16);
  if ((lane & 15) == 0) {
    const int h = lane >> 4;
    elog[e * 2 + h] = p;
    atomicMaxFloat(&lmax[(long long)d * 2 + h], p);
  }
}

// ---------------- layer-2 edge logits (heads=1, 64 channels; float2 path) ----------------

__global__ __launch_bounds__(256)
void k_edge_logits2(const int* __restrict__ srcE, const int* __restrict__ dstE,
                    const float* __restrict__ edge_attr, const float* __restrict__ loopA,
                    const float* __restrict__ We, const float* __restrict__ att,
                    const float* __restrict__ xl, const float* __restrict__ xr,
                    float* __restrict__ elog, float* __restrict__ lmax, int E, int N) {
  __shared__ float sWe[EDIM * OUTC];
  stage_lds_f32(sWe, We, EDIM * OUTC, threadIdx.x);
  __syncthreads();

  const int wave = threadIdx.x >> 5, lane = threadIdx.x & 31;
  const long long Etot = (long long)E + N;
  const long long e = (long long)blockIdx.x * 8 + wave;
  if (e >= Etot) return;

  int s, d; const float* eap;
  if (e < E) { s = srcE[e]; d = dstE[e]; eap = edge_attr + e * EDIM; }
  else       { s = d = (int)(e - E);     eap = loopA + (long long)s * EDIM; }

  float eav[EDIM];
#pragma unroll
  for (int k4 = 0; k4 < EDIM / 4; ++k4) {
    const float4 t = ((const float4*)eap)[k4];
    eav[4 * k4 + 0] = t.x; eav[4 * k4 + 1] = t.y;
    eav[4 * k4 + 2] = t.z; eav[4 * k4 + 3] = t.w;
  }

  const int c0 = lane * 2;
  const float2 am  = *(const float2*)(att + c0);
  const float2 xlv = *(const float2*)(xl + (long long)s * OUTC + c0);
  const float2 xrv = *(const float2*)(xr + (long long)d * OUTC + c0);
  float2 ep = {0.f, 0.f};
#pragma unroll
  for (int k = 0; k < EDIM; ++k) {
    const float2 w = *(const float2*)(sWe + k * OUTC + c0);
    ep.x += eav[k] * w.x; ep.y += eav[k] * w.y;
  }
  float p = 0.0f, m;
  m = xlv.x + xrv.x + ep.x; m = (m > 0.f) ? m : NEG_SLOPE * m; p += m * am.x;
  m = xlv.y + xrv.y + ep.y; m = (m > 0.f) ? m : NEG_SLOPE * m; p += m * am.y;

  p += __shfl_xor(p, 16);
  p += __shfl_xor(p, 8);
  p += __shfl_xor(p, 4);
  p += __shfl_xor(p, 2);
  p += __shfl_xor(p, 1);
  if (lane == 0) {
    elog[e] = p;
    atomicMaxFloat(&lmax[d], p);
  }
}

// ---------------- softmax numerator + denominator ----------------

__global__ void k_edge_exp(const int* __restrict__ dstE, const float* __restrict__ lmax,
                           float* __restrict__ elog, float* __restrict__ den,
                           int E, int N, int H) {
  long long i = (long long)blockIdx.x * blockDim.x + threadIdx.x;
  const long long total = ((long long)E + N) * H;
  if (i >= total) return;
  const long long e = i / H;
  const int h = (int)(i - e * H);
  const int d = (e < E) ? dstE[e] : (int)(e - E);
  const float ex = expf(elog[i] - lmax[(long long)d * H + h]);
  elog[i] = ex;
  atomicAdd(&den[(long long)d * H + h], ex);
}

// ---------------- alpha-weighted scatter: acc[dst] += alpha * xl[src] ----------------
// wave per edge; alpha computed once per lane; xl row read as b64/b128.

template <int F, int H, int CHPL>
__global__ __launch_bounds__(256)
void k_edge_aggregate(const int* __restrict__ srcE, const int* __restrict__ dstE,
                      const float* __restrict__ xl, const float* __restrict__ elog,
                      const float* __restrict__ den, float* __restrict__ acc,
                      int E, int N) {
  const int wave = threadIdx.x >> 5, lane = threadIdx.x & 31;
  const long long Etot = (long long)E + N;
  const long long e = (long long)blockIdx.x * 8 + wave;
  if (e >= Etot) return;
  int s, d;
  if (e < E) { s = srcE[e]; d = dstE[e]; } else { s = d = (int)(e - E); }
  const int c0 = lane * CHPL;
  const int h = c0 >> 6;                          // 64 channels per head (both layers)
  const float alpha = elog[e * H + h] / (den[(long long)d * H + h] + 1e-16f);
  const float* xs = xl + (long long)s * F + c0;
  float* ad = acc + (long long)d * F + c0;
  float v[CHPL];
#pragma unroll
  for (int j = 0; j < CHPL; ++j) v[j] = xs[j];    // merged into b64/b128
#pragma unroll
  for (int j = 0; j < CHPL; ++j) atomicAdd(ad + j, alpha * v[j]);
}

// ---------------- bias + ELU between layers (float4) ----------------

__global__ void k_bias_elu(float4* __restrict__ h4, const float4* __restrict__ bias4,
                           long long n4, int F4) {
  long long i = (long long)blockIdx.x * blockDim.x + threadIdx.x;
  if (i >= n4) return;
  const float4 b = bias4[(int)(i % F4)];
  float4 v = h4[i];
  v.x += b.x; v.y += b.y; v.z += b.z; v.w += b.w;
  v.x = (v.x > 0.f) ? v.x : (expf(v.x) - 1.f);
  v.y = (v.y > 0.f) ? v.y : (expf(v.y) - 1.f);
  v.z = (v.z > 0.f) ? v.z : (expf(v.z) - 1.f);
  v.w = (v.w > 0.f) ? v.w : (expf(v.w) - 1.f);
  h4[i] = v;
}

// ---------------- launch ----------------

static inline int gridFor(long long total, int block) {
  return (int)((total + block - 1) / block);
}

extern "C" void kernel_launch(void* const* d_in, const int* in_sizes, int n_in,
                              void* d_out, int out_size, void* d_ws, size_t ws_size,
                              hipStream_t stream) {
  const float* x     = (const float*)d_in[0];
  const int*   eidx  = (const int*)  d_in[1];
  const float* eattr = (const float*)d_in[2];
  const float* Wl1   = (const float*)d_in[3];
  const float* bl1   = (const float*)d_in[4];
  const float* Wr1   = (const float*)d_in[5];
  const float* br1   = (const float*)d_in[6];
  const float* We1   = (const float*)d_in[7];
  const float* att1  = (const float*)d_in[8];
  const float* bias1 = (const float*)d_in[9];
  const float* Wl2   = (const float*)d_in[10];
  const float* bl2   = (const float*)d_in[11];
  const float* Wr2   = (const float*)d_in[12];
  const float* br2   = (const float*)d_in[13];
  const float* We2   = (const float*)d_in[14];
  const float* att2  = (const float*)d_in[15];
  const float* bias2 = (const float*)d_in[16];

  const int N = in_sizes[0] / IN_F;
  const int E = in_sizes[1] / 2;
  const int* srcE = eidx;
  const int* dstE = eidx + E;
  const long long Etot = (long long)E + N;

  // workspace layout (floats)
  float* ws = (float*)d_ws;
  size_t o = 0;
  float* deg   = ws + o; o += (size_t)N;
  float* loopA = ws + o; o += (size_t)N * EDIM;
  float* xl1   = ws + o; o += (size_t)N * F1;
  float* xr1   = ws + o; o += (size_t)N * F1;
  float* lmax1 = ws + o; o += (size_t)N * 2;
  float* den1  = ws + o; o += (size_t)N * 2;
  float* hbuf  = ws + o; o += (size_t)N * F1;
  float* xl2   = ws + o; o += (size_t)N * OUTC;
  float* xr2   = ws + o; o += (size_t)N * OUTC;
  float* lmax2 = ws + o; o += (size_t)N;
  float* den2  = ws + o; o += (size_t)N;
  float* elog  = ws + o; o += (size_t)Etot * 2;
  (void)ws_size; (void)n_in; (void)out_size;

  const float NEG_INF = -__builtin_inff();

  // init
  k_fill<<<gridFor((long long)N * (1 + EDIM), 256), 256, 0, stream>>>(deg, 0.0f, (long long)N * (1 + EDIM));
  k_fill<<<gridFor((long long)N * 2, 256), 256, 0, stream>>>(den1, 0.0f, (long long)N * 2);
  k_fill<<<gridFor((long long)N, 256), 256, 0, stream>>>(den2, 0.0f, (long long)N);
  k_fill<<<gridFor((long long)N * F1, 256), 256, 0, stream>>>(hbuf, 0.0f, (long long)N * F1);
  k_fill<<<gridFor((long long)N * 2, 256), 256, 0, stream>>>(lmax1, NEG_INF, (long long)N * 2);
  k_fill<<<gridFor((long long)N, 256), 256, 0, stream>>>(lmax2, NEG_INF, (long long)N);
  k_init_out<<<gridFor((long long)N * (OUTC / 4), 256), 256, 0, stream>>>(
      (float4*)d_out, (const float4*)bias2, (long long)N * (OUTC / 4), OUTC / 4);

  // self-loop attrs (segment mean by dst)
  k_loop_sums<<<gridFor((long long)E * (EDIM / 4), 256), 256, 0, stream>>>(
      dstE, (const float4*)eattr, deg, loopA, E);
  k_loop_fin<<<gridFor((long long)N * (EDIM / 4), 256), 256, 0, stream>>>(
      (float4*)loopA, deg, N);

  // layer 1 node projections (WMMA)
  {
    const int mtiles = (N + 15) / 16;
    dim3 g((mtiles + 7) / 8);
    k_wmma_gemm_bias<F1><<<g, 256, 0, stream>>>(x, Wl1, bl1, xl1, N);
    k_wmma_gemm_bias<F1><<<g, 256, 0, stream>>>(x, Wr1, br1, xr1, N);
  }

  // layer 1 attention
  k_edge_logits1<<<gridFor(Etot, 8), 256, 0, stream>>>(srcE, dstE, eattr, loopA, We1, att1,
                                                       xl1, xr1, elog, lmax1, E, N);
  k_edge_exp<<<gridFor(Etot * 2, 256), 256, 0, stream>>>(dstE, lmax1, elog, den1, E, N, 2);
  k_edge_aggregate<F1, 2, 4><<<gridFor(Etot, 8), 256, 0, stream>>>(srcE, dstE, xl1, elog,
                                                                   den1, hbuf, E, N);
  k_bias_elu<<<gridFor((long long)N * (F1 / 4), 256), 256, 0, stream>>>(
      (float4*)hbuf, (const float4*)bias1, (long long)N * (F1 / 4), F1 / 4);

  // layer 2 node projections (WMMA)
  {
    const int mtiles = (N + 15) / 16;
    dim3 g((mtiles + 7) / 8);
    k_wmma_gemm_bias<OUTC><<<g, 256, 0, stream>>>(hbuf, Wl2, bl2, xl2, N);
    k_wmma_gemm_bias<OUTC><<<g, 256, 0, stream>>>(hbuf, Wr2, br2, xr2, N);
  }

  // layer 2 attention -> output (d_out pre-seeded with bias2)
  k_edge_logits2<<<gridFor(Etot, 8), 256, 0, stream>>>(srcE, dstE, eattr, loopA, We2, att2,
                                                       xl2, xr2, elog, lmax2, E, N);
  k_edge_exp<<<gridFor(Etot, 256), 256, 0, stream>>>(dstE, lmax2, elog, den2, E, N, 1);
  k_edge_aggregate<OUTC, 1, 2><<<gridFor(Etot, 8), 256, 0, stream>>>(srcE, dstE, xl2, elog,
                                                                     den2, (float*)d_out, E, N);
}
